// DNC_54348516163763
// MI455X (gfx1250) — matrix operand order repaired and checked by reading, not compiled, not run on hardware
//
#include <hip/hip_runtime.h>
#include <math.h>

// ---------------------------------------------------------------------------
// DNC step for MI455X (gfx1250, wave32).
// Dominant cost: link matrix (16x2048x2048 fp32 = 268MB). We read L exactly
// once and write link_new exactly once (~536MB => ~23us at 23.3 TB/s),
// fusing fwd_w (L.w), bwd_w (L^T.w) and the link update into one pass.
// Link-pass traffic is marked non-temporal (TH=NT): it has zero reuse and
// exceeds the 192MB L2, so keeping it out of L2 preserves cache for the
// reused operands (memory matrix, weights). Controller GEMMs (M=16 rows)
// use v_wmma_f32_16x16x32_f16.
// ---------------------------------------------------------------------------

#define B_   16
#define N_   2048
#define M_   64
#define R_   4
#define H_   512
#define CIN  320
#define COUT 535

typedef __attribute__((ext_vector_type(16))) _Float16 v16h;
typedef __attribute__((ext_vector_type(8)))  float    v8f;
typedef __attribute__((ext_vector_type(4)))  float    v4f;

// ---- workspace layout (float offsets) ----
#define WS_GATES   0            // 16*2048
#define WS_H       32768        // 16*512
#define WS_OUTP    40960        // 16*544 (stride COUT used, padded)
#define WS_PARSED  49664        // 16*544
#define WS_MN      58368        // 16*2048 memory row norms
#define WS_CW      91136        // 16*4*2048 content read weights
#define WS_CWW     222208       // 16*2048 content write weight
#define WS_ALLOC   254976       // 16*2048
#define WS_WW      287744       // 16*2048 write weights
#define WS_FWD     320512       // 16*4*2048 (atomically accumulated)
#define WS_BWD     451584       // 16*4*2048 (atomically accumulated)
#define WS_READS   582656       // 16*4*64

// ---- output layout (float offsets) ----
#define OUT_OUTPUT 0
#define OUT_MEM    1024
#define OUT_LINK   2098176
#define OUT_P      69207040
#define OUT_USAGE  69239808
#define OUT_READW  69272576

__device__ __forceinline__ float sigf(float x) { return 1.f / (1.f + __expf(-x)); }
__device__ __forceinline__ float softplusf(float x) {
    return fmaxf(x, 0.f) + log1pf(__expf(-fabsf(x)));
}

// ===========================================================================
// K1: gates = x @ W_ih^T + h @ W_hh^T + b_lstm   (16 x 2048), WMMA f16->f32
// One wave per 16-column tile. A rows = batch (M=16).
// A layout (16-bit, 16x32): lane l<16: row l, K in {k0+e, k0+16+e | e=0..7};
//                           lane l>=16: row l-16, K shifted by +8.
// B layout (32x16): lane l<16: col l, same K order as A; lanes 16-31: K shifted by +8.
// ===========================================================================
__device__ __forceinline__ float xval(const float* task, const float* prv,
                                      int b, int k) {
    return (k < 64) ? task[b * 64 + k] : prv[b * 256 + (k - 64)];
}

__global__ __launch_bounds__(32)
void k_gates(const float* __restrict__ task, const float* __restrict__ prv,
             const float* __restrict__ h_prev, const float* __restrict__ W_ih,
             const float* __restrict__ W_hh, const float* __restrict__ b_lstm,
             float* __restrict__ gates) {
    const int lane = threadIdx.x;
    const int l  = lane & 15;
    const int hi = lane >> 4;
    const int n0 = blockIdx.x * 16;
    const int sa = hi * 8;
    const int sb = hi * 16;
    v8f acc = {};

    for (int k0 = 0; k0 < CIN; k0 += 32) {        // x @ W_ih^T
        v16h a, bm;
        #pragma unroll
        for (int e = 0; e < 8; ++e) {
            a[e]     = (_Float16)xval(task, prv, l, k0 + sa + e);
            a[8 + e] = (_Float16)xval(task, prv, l, k0 + 16 + sa + e);
        }
        const float* wr = W_ih + (n0 + l) * CIN + k0 + sa;
        #pragma unroll
        for (int e = 0; e < 16; ++e) bm[e] = (_Float16)wr[e < 8 ? e : e + 8];
        acc = __builtin_amdgcn_wmma_f32_16x16x32_f16(false, a, false, bm,
                                                     (short)0, acc, false, false);
    }
    for (int k0 = 0; k0 < H_; k0 += 32) {         // h @ W_hh^T
        v16h a, bm;
        const float* hr = h_prev + l * H_;
        #pragma unroll
        for (int e = 0; e < 8; ++e) {
            a[e]     = (_Float16)hr[k0 + sa + e];
            a[8 + e] = (_Float16)hr[k0 + 16 + sa + e];
        }
        const float* wr = W_hh + (n0 + l) * H_ + k0 + sa;
        #pragma unroll
        for (int e = 0; e < 16; ++e) bm[e] = (_Float16)wr[e < 8 ? e : e + 8];
        acc = __builtin_amdgcn_wmma_f32_16x16x32_f16(false, a, false, bm,
                                                     (short)0, acc, false, false);
    }
    const float bias = b_lstm[n0 + l];
    #pragma unroll
    for (int v = 0; v < 8; ++v) {                 // D: VGPR v -> row v (+8 for hi)
        const int b = v + hi * 8;
        gates[b * 2048 + n0 + l] = acc[v] + bias;
    }
}

// ===========================================================================
// K2: LSTM pointwise: c = sig(gf)*c_prev + sig(gi)*tanh(gg); h = sig(go)*tanh(c)
// ===========================================================================
__global__ __launch_bounds__(256)
void k_lstm(const float* __restrict__ gates, const float* __restrict__ c_prev,
            float* __restrict__ h) {
    const int idx = blockIdx.x * blockDim.x + threadIdx.x;   // 16*512
    if (idx >= B_ * H_) return;
    const int b = idx >> 9, u = idx & 511;
    const float* g = gates + b * 2048;
    const float gi = g[u], gf = g[512 + u], gg = g[1024 + u], go = g[1536 + u];
    const float c = sigf(gf) * c_prev[idx] + sigf(gi) * tanhf(gg);
    h[idx] = sigf(go) * tanhf(c);
}

// ===========================================================================
// K3: out = h @ W_out^T + b_out  (16 x 535), WMMA with column guard
// ===========================================================================
__global__ __launch_bounds__(32)
void k_outproj(const float* __restrict__ h, const float* __restrict__ W_out,
               const float* __restrict__ b_out, float* __restrict__ outp) {
    const int lane = threadIdx.x;
    const int l  = lane & 15;
    const int hi = lane >> 4;
    const int n0 = blockIdx.x * 16;
    const int row = n0 + l;                 // output column (0..534)
    const bool ok = row < COUT;
    const int sa = hi * 8, sb = hi * 16;
    v8f acc = {};

    for (int k0 = 0; k0 < H_; k0 += 32) {
        v16h a, bm;
        const float* hr = h + l * H_;
        #pragma unroll
        for (int e = 0; e < 8; ++e) {
            a[e]     = (_Float16)hr[k0 + sa + e];
            a[8 + e] = (_Float16)hr[k0 + 16 + sa + e];
        }
        const float* wr = W_out + row * H_ + k0 + sa;
        #pragma unroll
        for (int e = 0; e < 16; ++e) bm[e] = ok ? (_Float16)wr[e < 8 ? e : e + 8] : (_Float16)0.f;
        acc = __builtin_amdgcn_wmma_f32_16x16x32_f16(false, a, false, bm,
                                                     (short)0, acc, false, false);
    }
    if (ok) {
        const float bias = b_out[row];
        #pragma unroll
        for (int v = 0; v < 8; ++v)
            outp[(v + hi * 8) * COUT + row] = acc[v] + bias;
    }
}

// ===========================================================================
// K4: parse controller outputs (sigmoid / one_plus / read-mode softmax)
// ===========================================================================
__global__ __launch_bounds__(544)
void k_parse(const float* __restrict__ o, float* __restrict__ p) {
    const int b = blockIdx.x, i = threadIdx.x;
    if (i >= COUT) return;
    const float* ob = o + b * COUT;
    float* pb = p + b * COUT;
    const float x = ob[i];
    if (i < 320)        pb[i] = sigf(x);                 // task_output, read_keys
    else if (i < 324)   pb[i] = 1.f + softplusf(x);      // read_betas
    else if (i < 392)   pb[i] = sigf(x);                 // free_gates, write_key
    else if (i == 392)  pb[i] = 1.f + softplusf(x);      // write_beta
    else if (i < 523)   pb[i] = sigf(x);                 // erase, write_vec, gates
    else if (((i - 523) % 3) == 0) {                     // read_modes softmax(3)
        const float a = ob[i], c = ob[i + 1], d = ob[i + 2];
        const float mx = fmaxf(a, fmaxf(c, d));
        const float ea = __expf(a - mx), eb = __expf(c - mx), ec = __expf(d - mx);
        const float inv = 1.f / (ea + eb + ec);
        pb[i] = ea * inv; pb[i + 1] = eb * inv; pb[i + 2] = ec * inv;
    }
}

// ===========================================================================
// K5: memory row norms
// ===========================================================================
__global__ __launch_bounds__(256)
void k_norms(const float* __restrict__ mem, float* __restrict__ mn) {
    const int idx = blockIdx.x * blockDim.x + threadIdx.x;   // 16*2048
    if (idx >= B_ * N_) return;
    const float* r = mem + (size_t)idx * M_;
    float s = 0.f;
    for (int m = 0; m < M_; m += 4) {
        const v4f v = *(const v4f*)(r + m);
        s += v.x * v.x + v.y * v.y + v.z * v.z + v.w * v.w;
    }
    mn[idx] = sqrtf(s);
}

// ===========================================================================
// K6: cosine content addressing + softmax over N. blockIdx.x=key(0..4: 4 read
// keys + write key), blockIdx.y=batch.
// ===========================================================================
__global__ __launch_bounds__(256)
void k_content(const float* __restrict__ mem, const float* __restrict__ mn,
               const float* __restrict__ parsed,
               float* __restrict__ cw, float* __restrict__ cww) {
    const int key = blockIdx.x, b = blockIdx.y, tid = threadIdx.x;
    __shared__ float ksh[M_];
    __shared__ float sc[N_];
    __shared__ float red[256];
    const float* pr = parsed + b * COUT;
    const float beta = (key < 4) ? pr[320 + key] : pr[392];
    if (tid < M_) ksh[tid] = (key < 4) ? pr[64 + key * 64 + tid] : pr[328 + tid];
    __syncthreads();
    float kn = 0.f;
    for (int m = 0; m < M_; ++m) kn += ksh[m] * ksh[m];
    kn = sqrtf(kn);

    for (int n = tid; n < N_; n += 256) {
        const float* row = mem + ((size_t)b * N_ + n) * M_;
        float dot = 0.f;
        for (int m = 0; m < M_; m += 4) {
            const v4f v = *(const v4f*)(row + m);
            dot += v.x * ksh[m] + v.y * ksh[m + 1] + v.z * ksh[m + 2] + v.w * ksh[m + 3];
        }
        sc[n] = beta * (dot / (kn * mn[b * N_ + n] + 1e-8f));
    }
    __syncthreads();
    float mx = -1e30f;
    for (int n = tid; n < N_; n += 256) mx = fmaxf(mx, sc[n]);
    red[tid] = mx; __syncthreads();
    for (int s = 128; s > 0; s >>= 1) {
        if (tid < s) red[tid] = fmaxf(red[tid], red[tid + s]);
        __syncthreads();
    }
    const float mxv = red[0]; __syncthreads();
    float sum = 0.f;
    for (int n = tid; n < N_; n += 256) { const float e = __expf(sc[n] - mxv); sc[n] = e; sum += e; }
    red[tid] = sum; __syncthreads();
    for (int s = 128; s > 0; s >>= 1) {
        if (tid < s) red[tid] += red[tid + s];
        __syncthreads();
    }
    const float inv = 1.f / red[0];
    for (int n = tid; n < N_; n += 256) {
        const float w = sc[n] * inv;
        if (key < 4) cw[((b * R_ + key) * N_) + n] = w;
        else         cww[b * N_ + n] = w;
    }
}

// ===========================================================================
// K7: usage + allocation (bitonic sort + multiplicative scan in LDS), per batch
// ===========================================================================
__global__ __launch_bounds__(1024)
void k_usage_alloc(const float* __restrict__ parsed, const float* __restrict__ prw,
                   const float* __restrict__ pu, const float* __restrict__ pww,
                   float* __restrict__ out, float* __restrict__ alloc) {
    const int b = blockIdx.x, tid = threadIdx.x;
    __shared__ float su[N_];    // sort keys / scan buffer
    __shared__ float uso[N_];   // sorted usage copy
    __shared__ float t2[N_];
    __shared__ int   si[N_];
    const float* pr = parsed + b * COUT;
    const float fg0 = pr[324], fg1 = pr[325], fg2 = pr[326], fg3 = pr[327];

    for (int n = tid; n < N_; n += 1024) {
        float psi = (1.f - fg0 * prw[(b * R_ + 0) * N_ + n]) *
                    (1.f - fg1 * prw[(b * R_ + 1) * N_ + n]) *
                    (1.f - fg2 * prw[(b * R_ + 2) * N_ + n]) *
                    (1.f - fg3 * prw[(b * R_ + 3) * N_ + n]);
        const float u0 = pu[b * N_ + n], w0 = pww[b * N_ + n];
        const float u = (u0 + w0 - u0 * w0) * psi;
        out[OUT_USAGE + b * N_ + n] = u;
        su[n] = u; si[n] = n;
    }
    __syncthreads();
    // bitonic sort ascending
    for (int k = 2; k <= N_; k <<= 1) {
        for (int j = k >> 1; j > 0; j >>= 1) {
            for (int i = tid; i < N_; i += 1024) {
                const int ixj = i ^ j;
                if (ixj > i) {
                    const bool up = ((i & k) == 0);
                    const float a = su[i], c = su[ixj];
                    if ((a > c) == up) {
                        su[i] = c; su[ixj] = a;
                        const int ti = si[i]; si[i] = si[ixj]; si[ixj] = ti;
                    }
                }
            }
            __syncthreads();
        }
    }
    for (int i = tid; i < N_; i += 1024) uso[i] = su[i];
    __syncthreads();
    // inclusive multiplicative scan (Hillis-Steele)
    for (int off = 1; off < N_; off <<= 1) {
        for (int i = tid; i < N_; i += 1024)
            t2[i] = (i >= off) ? su[i - off] * su[i] : su[i];
        __syncthreads();
        for (int i = tid; i < N_; i += 1024) su[i] = t2[i];
        __syncthreads();
    }
    for (int i = tid; i < N_; i += 1024) {
        const float excl = (i == 0) ? 1.f : su[i - 1];
        alloc[b * N_ + si[i]] = (1.f - uso[i]) * excl;
    }
}

// ===========================================================================
// K8: write weights, p_new, memory_new (per batch). memory_new store is NT.
// ===========================================================================
__global__ __launch_bounds__(256)
void k_write(const float* __restrict__ parsed, const float* __restrict__ alloc,
             const float* __restrict__ cww, const float* __restrict__ prec,
             const float* __restrict__ mem, float* __restrict__ ww,
             float* __restrict__ out) {
    const int b = blockIdx.x, tid = threadIdx.x;
    __shared__ float ev[M_], wv[M_], red[256];
    const float* pr = parsed + b * COUT;
    const float ag = pr[521], wg = pr[522];
    if (tid < M_) { ev[tid] = pr[393 + tid]; wv[tid] = pr[457 + tid]; }
    float s = 0.f;
    for (int n = tid; n < N_; n += 256) {
        const float w = wg * (ag * alloc[b * N_ + n] + (1.f - ag) * cww[b * N_ + n]);
        ww[b * N_ + n] = w; s += w;
    }
    red[tid] = s; __syncthreads();
    for (int k = 128; k > 0; k >>= 1) {
        if (tid < k) red[tid] += red[tid + k];
        __syncthreads();
    }
    const float S = red[0];
    for (int n = tid; n < N_; n += 256) {
        const float w = ww[b * N_ + n];
        out[OUT_P + b * N_ + n] = (1.f - S) * prec[b * N_ + n] + w;
        const float* mr = mem + ((size_t)b * N_ + n) * M_;
        float* orow = out + OUT_MEM + ((size_t)b * N_ + n) * M_;
        for (int m = 0; m < M_; m += 4) {
            const v4f v = *(const v4f*)(mr + m);
            v4f o;
            o.x = v.x * (1.f - w * ev[m])     + w * wv[m];
            o.y = v.y * (1.f - w * ev[m + 1]) + w * wv[m + 1];
            o.z = v.z * (1.f - w * ev[m + 2]) + w * wv[m + 2];
            o.w = v.w * (1.f - w * ev[m + 3]) + w * wv[m + 3];
            __builtin_nontemporal_store(o, (v4f*)(orow + m));
        }
    }
}

// ===========================================================================
// K9: FUSED link pass — reads L once (NT load: zero reuse, > L2 capacity),
// writes link_new once (NT store), accumulates fwd_w (per-row wave shfl
// reduce -> LDS atomics -> one global flush) and bwd_w (per-column
// registers -> global atomics).
// grid: (8 col-tiles of 256, 4 row-segments of 512, 16 batches); 256 thr.
// ===========================================================================
__global__ __launch_bounds__(256)
void k_link(const float* __restrict__ L, const float* __restrict__ ww,
            const float* __restrict__ prw, float* __restrict__ out,
            float* __restrict__ fwd, float* __restrict__ bwd) {
    const int jt = blockIdx.x, seg = blockIdx.y, b = blockIdx.z;
    const int tid = threadIdx.x;
    const int i0 = seg * 512;
    const int j  = jt * 256 + tid;
    __shared__ float wi[R_][512];   // prev_read_weights over this row segment
    __shared__ float wwi[512];      // write weights over this row segment
    __shared__ float fsh[R_][512];  // fwd partials for this block

    for (int t = tid; t < 512; t += 256) {
        wwi[t] = ww[b * N_ + i0 + t];
        #pragma unroll
        for (int r = 0; r < R_; ++r) {
            wi[r][t]  = prw[(b * R_ + r) * N_ + i0 + t];
            fsh[r][t] = 0.f;
        }
    }
    const float wwj = ww[b * N_ + j];
    const float pj  = out[OUT_P + b * N_ + j];
    float wrj[R_], bacc[R_];
    #pragma unroll
    for (int r = 0; r < R_; ++r) { wrj[r] = prw[(b * R_ + r) * N_ + j]; bacc[r] = 0.f; }
    __syncthreads();

    const float* Lb = L   + (size_t)b * N_ * N_;
    float*       Ob = out + OUT_LINK + (size_t)b * N_ * N_;

    for (int ic = 0; ic < 512; ic += 4) {
        __builtin_prefetch(Lb + (size_t)(i0 + ic + 4) * N_ + j, 0, 1);
        float prod[4][R_];
        #pragma unroll
        for (int c = 0; c < 4; ++c) {
            const int il = ic + c;
            const int i  = i0 + il;
            const float l = __builtin_nontemporal_load(Lb + (size_t)i * N_ + j);
            const float wr = wwi[il];
            const float lnew = (i == j) ? 0.f : (1.f - wr - wwj) * l + wr * pj;
            __builtin_nontemporal_store(lnew, Ob + (size_t)i * N_ + j);
            #pragma unroll
            for (int r = 0; r < R_; ++r) {
                bacc[r] += l * wi[r][il];     // bwd: column accumulator
                prod[c][r] = l * wrj[r];      // fwd: per-row partial product
            }
        }
        #pragma unroll
        for (int c = 0; c < 4; ++c) {
            #pragma unroll
            for (int r = 0; r < R_; ++r) {
                float v = prod[c][r];
                v += __shfl_xor(v, 1, 32);
                v += __shfl_xor(v, 2, 32);
                v += __shfl_xor(v, 4, 32);
                v += __shfl_xor(v, 8, 32);
                v += __shfl_xor(v, 16, 32);
                if ((tid & 31) == 0) atomicAdd(&fsh[r][ic + c], v);
            }
        }
    }
    __syncthreads();
    for (int t = tid; t < R_ * 512; t += 256) {
        const int r = t >> 9, il = t & 511;
        atomicAdd(&fwd[(b * R_ + r) * N_ + i0 + il], fsh[r][il]);
    }
    #pragma unroll
    for (int r = 0; r < R_; ++r)
        atomicAdd(&bwd[(b * R_ + r) * N_ + j], bacc[r]);
}

// ===========================================================================
// K10: read weights = m0*bwd + m1*content + m2*fwd
// ===========================================================================
__global__ __launch_bounds__(256)
void k_readw(const float* __restrict__ parsed, const float* __restrict__ cw,
             const float* __restrict__ fwd, const float* __restrict__ bwd,
             float* __restrict__ out) {
    const int idx = blockIdx.x * blockDim.x + threadIdx.x;   // B*R*N
    if (idx >= B_ * R_ * N_) return;
    const int r = (idx >> 11) & 3, b = idx >> 13;
    const float* pr = parsed + b * COUT + 523 + r * 3;
    out[OUT_READW + idx] = pr[0] * bwd[idx] + pr[1] * cw[idx] + pr[2] * fwd[idx];
}

// ===========================================================================
// K11: reads = read_w @ memory  (per (b,r): 64-wide over 2048 rows)
// ===========================================================================
__global__ __launch_bounds__(256)
void k_reads(const float* __restrict__ out, const float* __restrict__ mem,
             float* __restrict__ reads) {
    const int br = blockIdx.x;                 // 0..63
    const int b = br >> 2, r = br & 3;
    const int tid = threadIdx.x, m = tid & 63, g = tid >> 6;
    __shared__ float red[4][M_];
    float acc = 0.f;
    const float* rw = out + OUT_READW + (size_t)(b * R_ + r) * N_;
    for (int n = g; n < N_; n += 4)
        acc += rw[n] * mem[((size_t)b * N_ + n) * M_ + m];
    red[g][m] = acc;
    __syncthreads();
    if (g == 0)
        reads[(b * R_ + r) * M_ + m] = red[0][m] + red[1][m] + red[2][m] + red[3][m];
}

// ===========================================================================
// K12: final head: sigmoid(sigmoid(reads@W_read^T + b_read) + task_output)
// ===========================================================================
__global__ __launch_bounds__(1024)
void k_final(const float* __restrict__ reads, const float* __restrict__ W_read,
             const float* __restrict__ b_read, const float* __restrict__ parsed,
             float* __restrict__ out) {
    const int tid = threadIdx.x;               // 16*64
    const int b = tid >> 6, t = tid & 63;
    float acc = b_read[t];
    const float* rr = reads + b * (R_ * M_);
    const float* wr = W_read + t * (R_ * M_);
    for (int k = 0; k < R_ * M_; ++k) acc += rr[k] * wr[k];
    out[OUT_OUTPUT + b * 64 + t] = sigf(sigf(acc) + parsed[b * COUT + t]);
}

// ===========================================================================
extern "C" void kernel_launch(void* const* d_in, const int* in_sizes, int n_in,
                              void* d_out, int out_size, void* d_ws, size_t ws_size,
                              hipStream_t stream) {
    const float* task   = (const float*)d_in[0];
    const float* memory = (const float*)d_in[1];
    const float* links  = (const float*)d_in[2];
    const float* prv    = (const float*)d_in[3];
    const float* prw    = (const float*)d_in[4];
    const float* pu     = (const float*)d_in[5];
    const float* pww    = (const float*)d_in[6];
    const float* prec   = (const float*)d_in[7];
    const float* h_prev = (const float*)d_in[8];
    const float* c_prev = (const float*)d_in[9];
    const float* W_ih   = (const float*)d_in[10];
    const float* W_hh   = (const float*)d_in[11];
    const float* b_lstm = (const float*)d_in[12];
    const float* W_out  = (const float*)d_in[13];
    const float* b_out  = (const float*)d_in[14];
    const float* W_read = (const float*)d_in[15];
    const float* b_read = (const float*)d_in[16];
    float* out = (float*)d_out;
    float* ws  = (float*)d_ws;

    // zero fwd/bwd accumulators (contiguous region)
    (void)hipMemsetAsync(ws + WS_FWD, 0,
                         (size_t)(WS_BWD + B_ * R_ * N_ - WS_FWD) * sizeof(float),
                         stream);

    k_gates  <<<128, 32, 0, stream>>>(task, prv, h_prev, W_ih, W_hh, b_lstm,
                                      ws + WS_GATES);
    k_lstm   <<<32, 256, 0, stream>>>(ws + WS_GATES, c_prev, ws + WS_H);
    k_outproj<<<34, 32, 0, stream>>>(ws + WS_H, W_out, b_out, ws + WS_OUTP);
    k_parse  <<<16, 544, 0, stream>>>(ws + WS_OUTP, ws + WS_PARSED);
    k_norms  <<<128, 256, 0, stream>>>(memory, ws + WS_MN);
    k_content<<<dim3(5, 16), 256, 0, stream>>>(memory, ws + WS_MN, ws + WS_PARSED,
                                               ws + WS_CW, ws + WS_CWW);
    k_usage_alloc<<<16, 1024, 0, stream>>>(ws + WS_PARSED, prw, pu, pww,
                                           out, ws + WS_ALLOC);
    k_write  <<<16, 256, 0, stream>>>(ws + WS_PARSED, ws + WS_ALLOC, ws + WS_CWW,
                                      prec, memory, ws + WS_WW, out);
    k_link   <<<dim3(8, 4, 16), 256, 0, stream>>>(links, ws + WS_WW, prw, out,
                                                  ws + WS_FWD, ws + WS_BWD);
    k_readw  <<<512, 256, 0, stream>>>(ws + WS_PARSED, ws + WS_CW,
                                       ws + WS_FWD, ws + WS_BWD, out);
    k_reads  <<<64, 256, 0, stream>>>(out, memory, ws + WS_READS);
    k_final  <<<1, 1024, 0, stream>>>(ws + WS_READS, W_read, b_read,
                                      ws + WS_PARSED, out);
}
